// SinenetLayer_11527692222698
// MI455X (gfx1250) — matrix operand order repaired and checked by reading, not compiled, not run on hardware
//
#include <hip/hip_runtime.h>
#include <hip/hip_fp16.h>
#include <math.h>

typedef __attribute__((ext_vector_type(16))) _Float16 v16h;
typedef __attribute__((ext_vector_type(8)))  float    v8f;

#define S_  25
#define B_  64
#define T_  640
#define D_  128
#define K_  64                   // 32 sin + 32 cos basis slots

#define TGROUPS 8                // t-stride groups per block
#define TITER  (T_ / TGROUPS)    // 80 recurrence steps per thread

__device__ __constant__ float kTWAV      = 1.0f / 16000.0f;
__device__ __constant__ float kLOGF_MEAN = 5.02654f;
__device__ __constant__ float kLOGF_STD  = 0.373288f;
__device__ __constant__ float kTWO_PI    = 6.28318530717958647692f;

// ---------------------------------------------------------------------------
// Stage 0: transposed mixing matrix Wt[d][j] (f16).
//   h[s,b,d] = a[d]*( cos(phi_d)*Ssum_i(d) + sin(phi_d)*Csum_i(d) )
// so W[j,d] = a*cos(phi) at j=i(d), a*sin(phi) at j=32+i(d), else 0.
// Stored d-major so each WMMA B-lane reads 16 contiguous halves.
// ---------------------------------------------------------------------------
__global__ void w_init_kernel(const float* __restrict__ a,
                              const float* __restrict__ phi,
                              _Float16* __restrict__ Wt) {
  int d = threadIdx.x;            // 128 threads
  float ad = a[d];
  float sp, cp;
  sincosf(phi[d], &sp, &cp);
  int ig = d >> 2;                // harmonic group 0..31
  float wc = ad * cp;             // weight on sin(i*theta)
  float ws = ad * sp;             // weight on cos(i*theta)
#pragma unroll
  for (int j = 0; j < K_; ++j) {
    float v = (j == ig) ? wc : ((j == 32 + ig) ? ws : 0.0f);
    Wt[d * K_ + j] = (_Float16)v;
  }
}

// ---------------------------------------------------------------------------
// Stage 1: SC[sb][j]   (j<32: sum_t sin((j+1)theta_t) x_t,
//                       j>=32: sum_t cos((j-31)theta_t) x_t)
// theta_t = 2*pi*f*(t*T_WAV - tau).
// One block per (s,b); 256 threads = 32 harmonics x 8 t-groups.
// The angle is an arithmetic progression in t, so each thread does ONE
// accurate sincos pair (initial phase + step) and then a 6-FMA Givens
// rotation per sample — no transcendentals in the hot loop, and each
// rotation feeds BOTH the sin- and cos- accumulators.
// x row (2.5KB) is staged through LDS once; 32-way broadcast ds_loads after.
// ---------------------------------------------------------------------------
__global__ __launch_bounds__(256)
void sc_kernel(const float* __restrict__ x,
               const float* __restrict__ nlf,
               const float* __restrict__ tau,
               _Float16* __restrict__ SC) {
  __shared__ float xs[T_];
  __shared__ float redS[256];
  __shared__ float redC[256];

  const int sb  = blockIdx.x;
  const int tid = threadIdx.x;
  const int h   = tid & 31;       // harmonic slot -> i = h+1
  const int g   = tid >> 5;       // t-stride group 0..7

  const float* __restrict__ xp = x + (size_t)sb * T_;
#pragma unroll
  for (int t = tid; t < T_; t += 256) xs[t] = xp[t];

  const float f      = expf(nlf[sb] * kLOGF_STD + kLOGF_MEAN);
  const float tau_sb = tau[sb];
  const float w      = kTWO_PI * f * (float)(h + 1);   // angular rate of harmonic

  // Exact trig only here: initial phase (large arg, OCML range reduction)
  // and the per-step rotation angle.
  float sv, cv, sd, cd;
  sincosf(w * ((float)g * kTWAV - tau_sb), &sv, &cv);
  sincosf(w * ((float)TGROUPS * kTWAV),    &sd, &cd);

  __syncthreads();

  float accS = 0.0f, accC = 0.0f;
#pragma unroll 4
  for (int t = g; t < T_; t += TGROUPS) {
    float xv = xs[t];
    accS = fmaf(sv, xv, accS);
    accC = fmaf(cv, xv, accC);
    float sn = fmaf(sv, cd,  cv * sd);   // sin(ang+d)
    float cn = fmaf(cv, cd, -sv * sd);   // cos(ang+d)
    sv = sn; cv = cn;
  }

  redS[tid] = accS;
  redC[tid] = accC;
  __syncthreads();

  // 64 result slots: j<32 from redS, j>=32 from redC; sum 8 t-group partials.
  if (tid < K_) {
    const float* r  = (tid < 32) ? redS : redC;
    const int    hh = tid & 31;
    float s = 0.0f;
#pragma unroll
    for (int gg = 0; gg < TGROUPS; ++gg) s += r[hh + 32 * gg];
    SC[(size_t)sb * K_ + tid] = (_Float16)s;
  }
}

// ---------------------------------------------------------------------------
// Stage 2: out[m][d] = sum_j SC[m][j] * W[j][d]  as 1600x128x64 WMMA GEMM.
// One wave32 per 16x16 tile; 100 blocks x 8 waves. K=64 -> two 16x16x32 steps.
// Fragment packing used here (wave32; A and B share a contiguous K order):
//  A 16-bit 16x32: lane<16 -> M=lane, halves[0..15]=K 0..15
//                  lane>=16 -> M=lane-16, K 16..31
//  B 16-bit 32x16: lane holds column N=lane&15, 16 consecutive K
//                  (lo lanes K 0..15, hi lanes K 16..31) -> Wt is d-major.
//  C/D f32 16x16:  element r -> M = r + 8*(lane>=16), N = lane&15.
// ---------------------------------------------------------------------------
__global__ __launch_bounds__(256)
void gemm_kernel(const _Float16* __restrict__ SC,
                 const _Float16* __restrict__ Wt,
                 float* __restrict__ out) {
  const int lane  = threadIdx.x & 31;
  const int wv    = threadIdx.x >> 5;   // N-tile 0..7
  const int m0    = blockIdx.x << 4;    // M-tile base row
  const int d0    = wv << 4;            // N-tile base col
  const int mlane = lane & 15;
  const int hi    = lane >> 4;          // 0: lanes 0-15, 1: lanes 16-31

  v8f c = {};
#pragma unroll
  for (int kh = 0; kh < K_; kh += 32) {
    // --- A fragment (SC row-major, rows of 64 halves = 128B aligned) ---
    const _Float16* arow = SC + (size_t)(m0 + mlane) * K_ + kh + hi * 16;
    v16h af;
#pragma unroll
    for (int e = 0; e < 8; ++e) {
      af[e]     = arow[e];        // K = kh + hi*16 + e
      af[e + 8] = arow[e + 8];    // K = kh + hi*16 + 8 + e
    }
    // --- B fragment (Wt d-major: 16 contiguous halves per lane) ---
    const _Float16* bcol = Wt + (size_t)(d0 + mlane) * K_ + kh + hi * 16;
    v16h bf;
#pragma unroll
    for (int e = 0; e < 16; ++e) bf[e] = bcol[e];

    c = __builtin_amdgcn_wmma_f32_16x16x32_f16(
        /*neg_a=*/false, af, /*neg_b=*/false, bf,
        /*c_mod=*/(short)0, c, /*reuse_a=*/false, /*reuse_b=*/false);
  }

#pragma unroll
  for (int r = 0; r < 8; ++r) {
    int m = m0 + r + hi * 8;
    out[(size_t)m * D_ + d0 + mlane] = c[r];
  }
}

// ---------------------------------------------------------------------------
extern "C" void kernel_launch(void* const* d_in, const int* in_sizes, int n_in,
                              void* d_out, int out_size, void* d_ws, size_t ws_size,
                              hipStream_t stream) {
  const float* x   = (const float*)d_in[0];   // (S,B,1,T)
  const float* nlf = (const float*)d_in[1];   // (S,B,1,1)
  const float* tau = (const float*)d_in[2];   // (S,B,1,1)
  const float* a   = (const float*)d_in[3];   // (D,)
  const float* phi = (const float*)d_in[4];   // (D,)
  float* out = (float*)d_out;                 // (S,B,D) f32

  _Float16* Wt = (_Float16*)d_ws;                                   // 128*64 f16
  _Float16* SC = (_Float16*)((char*)d_ws + (size_t)D_ * K_ * 2);    // 1600*64 f16

  w_init_kernel<<<1, D_, 0, stream>>>(a, phi, Wt);
  sc_kernel<<<S_ * B_, 256, 0, stream>>>(x, nlf, tau, SC);
  gemm_kernel<<<(S_ * B_) / 16, 256, 0, stream>>>(SC, Wt, out);
}
